// RetrievalNet_57543971832710
// MI455X (gfx1250) — matrix operand order repaired and checked by reading, not compiled, not run on hardware
//
#include <hip/hip_runtime.h>
#include <hip/hip_bf16.h>
#include <stdint.h>

// ---------------------------------------------------------------------------
// RetrievalNet on MI455X (gfx1250): bf16 WMMA GEMM pipeline, fp32 accumulate,
// async-to-LDS double-buffered tile staging (ASYNCcnt path).
// ---------------------------------------------------------------------------

typedef __attribute__((ext_vector_type(8)))  float  v8f;
typedef __attribute__((ext_vector_type(16))) __bf16 v16bf;
typedef __attribute__((ext_vector_type(8)))  __bf16 v8bf;
typedef __attribute__((ext_vector_type(4)))  int    v4i;

#define DEV __device__ __forceinline__

static constexpr int B_   = 2048;
static constexpr int T_   = 32;
static constexpr int DV   = 2048;
static constexpr int DL   = 512;
static constexpr int H_   = 1024;
static constexpr int E_   = 1024;
static constexpr int VIS1 = 2048;
static constexpr int VIS2 = 1024;
static constexpr int KP1  = 2080;        // DV+2=2050 padded to multiple of 32
static constexpr int KC   = DL + H_;     // 1536 fused LSTM GEMM K
static constexpr int G4H  = 4 * H_;      // 4096
static constexpr float MARGIN   = 0.1f;
static constexpr float LW_INTER = 1.0f;
static constexpr float LW_INTRA = 1.0f;

DEV unsigned short f2bf_bits(float x) {
  unsigned u = __builtin_bit_cast(unsigned, x);
  unsigned r = (u + 0x7FFFu + ((u >> 16) & 1u)) >> 16;   // RNE
  return (unsigned short)r;
}
DEV float sigm(float x) { return 1.0f / (1.0f + __expf(-x)); }

// ---- CDNA5 async global->LDS copy (16B per lane), ASYNCcnt tracked --------
DEV void async_cp16(const unsigned short* g, unsigned short* l) {
#if defined(__has_builtin) && __has_builtin(__builtin_amdgcn_global_load_async_to_lds_b128)
  __builtin_amdgcn_global_load_async_to_lds_b128(
      (__attribute__((address_space(1))) v4i*)(uintptr_t)g,
      (__attribute__((address_space(3))) v4i*)(unsigned)(uintptr_t)l,
      0, 0);
#else
  unsigned loff = (unsigned)(uintptr_t)l;    // LDS byte offset = addr[31:0]
  asm volatile("global_load_async_to_lds_b128 %0, %1, off"
               :: "v"(loff), "v"(g) : "memory");
#endif
}
DEV void wait_async0() {
#if defined(__has_builtin) && __has_builtin(__builtin_amdgcn_s_wait_asynccnt)
  __builtin_amdgcn_s_wait_asynccnt(0);
#else
  asm volatile("s_wait_asynccnt 0x0" ::: "memory");
#endif
}

// ---------------------------------------------------------------------------
// Generic bf16 GEMM:  C[M x N] = A[M x K] * W[N x K]^T (+bias0 +bias1) (+relu)
// 256 threads = 8 waves; wave computes 32x32 via 2x2 v_wmma_f32_16x16x32_bf16.
// Block tile: 128 (M, blockIdx.y) x 64 (N, blockIdx.x), K-step 32.
// A/B tiles staged into double-buffered LDS via global_load_async_to_lds_b128.
// ---------------------------------------------------------------------------
__global__ __launch_bounds__(256) void gemm_bf16_wmma(
    const unsigned short* __restrict__ A, int lda,
    const unsigned short* __restrict__ W, int ldb,
    const float* __restrict__ bias0, const float* __restrict__ bias1,
    float* __restrict__ Cf, unsigned short* __restrict__ Cbf,
    int N, int K, int relu)
{
  __shared__ __align__(64) unsigned short lAsh[2][128 * 32];   // 2 x 8KB
  __shared__ __align__(64) unsigned short lBsh[2][64 * 32];    // 2 x 4KB

  const int tid   = threadIdx.x;
  const int lane  = tid & 31;
  const int wave  = tid >> 5;
  const int mBlk  = blockIdx.y * 128;
  const int nBlk  = blockIdx.x * 64;
  const int mW    = (wave & 3) * 32;    // wave tile row within block
  const int nW    = (wave >> 2) * 32;   // wave tile col within block

  const int l15   = lane & 15;
  const int hi16  = (lane & 16) ? 1 : 0;
  const int kOffA = hi16 ? 16 : 0;      // A frag: lanes 16-31 hold K 16..31
  const int kOffB = hi16 ? 16 : 0;      // B frag: lanes 16-31 hold K 16..31

  // per-thread async-copy slots
  const int aRow = tid >> 1;            // 0..127
  const int aCol = (tid & 1) * 16;      // element offset 0 / 16
  const int bRow = tid >> 2;            // 0..63
  const int bCol = (tid & 3) * 8;       // element offset 0/8/16/24
  const unsigned short* gA = A + (size_t)(mBlk + aRow) * lda + aCol;
  const unsigned short* gB = W + (size_t)(nBlk + bRow) * ldb + bCol;

  auto issue = [&](int buf, int k) {
    unsigned short* lA = &lAsh[buf][aRow * 32 + aCol];
    async_cp16(gA + k,     lA);
    async_cp16(gA + k + 8, lA + 8);
    async_cp16(gB + k, &lBsh[buf][bRow * 32 + bCol]);
  };

  v8f c00 = {}, c01 = {}, c10 = {}, c11 = {};

  issue(0, 0);
  int buf = 0;
  for (int k = 0; k < K; k += 32) {
    wait_async0();        // my copies into `buf` are in LDS
    __syncthreads();      // everyone's copies visible; everyone done with buf^1
    if (k + 32 < K) issue(buf ^ 1, k + 32);

    const unsigned short* lA = lAsh[buf];
    const unsigned short* lB = lBsh[buf];

    const __bf16* pa0 = reinterpret_cast<const __bf16*>(lA + (mW + l15) * 32 + kOffA);
    const __bf16* pa1 = reinterpret_cast<const __bf16*>(lA + (mW + 16 + l15) * 32 + kOffA);
    v8bf a0l = *reinterpret_cast<const v8bf*>(pa0);
    v8bf a0h = *reinterpret_cast<const v8bf*>(pa0 + 8);
    v8bf a1l = *reinterpret_cast<const v8bf*>(pa1);
    v8bf a1h = *reinterpret_cast<const v8bf*>(pa1 + 8);
    v16bf b0 = *reinterpret_cast<const v16bf*>(lB + (nW + l15) * 32 + kOffB);
    v16bf b1 = *reinterpret_cast<const v16bf*>(lB + (nW + 16 + l15) * 32 + kOffB);
    v16bf a0, a1;
#pragma unroll
    for (int i = 0; i < 8; ++i) {
      a0[i] = a0l[i]; a0[i + 8] = a0h[i];
      a1[i] = a1l[i]; a1[i + 8] = a1h[i];
    }
    c00 = __builtin_amdgcn_wmma_f32_16x16x32_bf16(false, a0, false, b0, (short)0, c00, false, false);
    c01 = __builtin_amdgcn_wmma_f32_16x16x32_bf16(false, a0, false, b1, (short)0, c01, false, false);
    c10 = __builtin_amdgcn_wmma_f32_16x16x32_bf16(false, a1, false, b0, (short)0, c10, false, false);
    c11 = __builtin_amdgcn_wmma_f32_16x16x32_bf16(false, a1, false, b1, (short)0, c11, false, false);
    buf ^= 1;
  }

  // C/D layout: lanes 0-15 -> M = r, N = lane; lanes 16-31 -> M = r+8, N = lane-16
  const int rowOff = hi16 ? 8 : 0;
  const v8f accs[4] = { c00, c01, c10, c11 };
#pragma unroll
  for (int t = 0; t < 4; ++t) {
    const int tm = t >> 1, tn = t & 1;
    const int n0 = nBlk + nW + tn * 16 + l15;
    const int m0 = mBlk + mW + tm * 16 + rowOff;
    float bv = (bias0 ? bias0[n0] : 0.0f) + (bias1 ? bias1[n0] : 0.0f);
#pragma unroll
    for (int r = 0; r < 8; ++r) {
      float v = accs[t][r] + bv;
      if (relu) v = v > 0.0f ? v : 0.0f;
      size_t off = (size_t)(m0 + r) * N + n0;
      if (Cf)  Cf[off]  = v;
      if (Cbf) Cbf[off] = f2bf_bits(v);
    }
  }
}

// ------------------------- elementwise helper kernels ----------------------

__global__ void k_zero32(unsigned int* p, size_t n) {
  for (size_t i = (size_t)blockIdx.x * blockDim.x + threadIdx.x; i < n;
       i += (size_t)gridDim.x * blockDim.x) p[i] = 0u;
}

// fp32 [N x Ksrc] -> bf16 [N x Kdst], zero-padded along K
__global__ void k_convert_pad(unsigned short* dst, const float* src, int N, int Ksrc, int Kdst) {
  size_t total = (size_t)N * Kdst;
  for (size_t i = (size_t)blockIdx.x * blockDim.x + threadIdx.x; i < total;
       i += (size_t)gridDim.x * blockDim.x) {
    int n = (int)(i / Kdst), k = (int)(i % Kdst);
    dst[i] = (k < Ksrc) ? f2bf_bits(src[(size_t)n * Ksrc + k]) : (unsigned short)0;
  }
}

// fused LSTM weight: Wc[4H x (DL+H)] = [W_ih | W_hh] as bf16
__global__ void k_build_wc(unsigned short* dst, const float* wih, const float* whh) {
  size_t total = (size_t)G4H * KC;
  for (size_t i = (size_t)blockIdx.x * blockDim.x + threadIdx.x; i < total;
       i += (size_t)gridDim.x * blockDim.x) {
    int n = (int)(i / KC), k = (int)(i % KC);
    float v = (k < DL) ? wih[(size_t)n * DL + k] : whh[(size_t)n * H_ + (k - DL)];
    dst[i] = f2bf_bits(v);
  }
}

// image A: [B x KP1] = [feats | ts | 0-pad] as bf16
__global__ void k_build_imgA(unsigned short* dst, const float* feats, const float* ts) {
  size_t total = (size_t)B_ * KP1;
  for (size_t i = (size_t)blockIdx.x * blockDim.x + threadIdx.x; i < total;
       i += (size_t)gridDim.x * blockDim.x) {
    int b = (int)(i / KP1), k = (int)(i % KP1);
    float v = (k < DV) ? feats[(size_t)b * DV + k]
                       : ((k < DV + 2) ? ts[(size_t)b * 2 + (k - DV)] : 0.0f);
    dst[i] = f2bf_bits(v);
  }
}

// copy query[t] (B x DL fp32) into cols [0,DL) of A_lstm (B x KC bf16)
__global__ void k_query_to_A(unsigned short* Al, const float* qt) {
  size_t total = (size_t)B_ * DL;
  for (size_t i = (size_t)blockIdx.x * blockDim.x + threadIdx.x; i < total;
       i += (size_t)gridDim.x * blockDim.x) {
    int b = (int)(i / DL), k = (int)(i % DL);
    Al[(size_t)b * KC + k] = f2bf_bits(qt[i]);
  }
}

// LSTM cell: gates [B x 4H] order (i,f,g,o).  Writes:
//   h fp32 (unmasked), c fp32 (masked by cont_next), and masked bf16 h into
//   cols [DL, DL+H) of A_lstm to feed the next step's fused GEMM.
__global__ void k_lstm_cell(const float* __restrict__ gates, float* __restrict__ c,
                            float* __restrict__ h, unsigned short* __restrict__ Al,
                            const float* __restrict__ cont_next) {
  size_t total = (size_t)B_ * H_;
  for (size_t i = (size_t)blockIdx.x * blockDim.x + threadIdx.x; i < total;
       i += (size_t)gridDim.x * blockDim.x) {
    int b = (int)(i / H_), j = (int)(i % H_);
    const float* g = gates + (size_t)b * G4H;
    float ig = g[j], fg = g[H_ + j], gg = g[2 * H_ + j], og = g[3 * H_ + j];
    float cN = sigm(fg) * c[i] + sigm(ig) * tanhf(gg);
    float hN = sigm(og) * tanhf(cN);
    float ct = cont_next ? cont_next[b] : 1.0f;
    c[i] = cN * ct;
    h[i] = hN;
    Al[(size_t)b * KC + DL + j] = f2bf_bits(hN * ct);
  }
}

// per-row squared distances + hinge margins (deterministic block reduction)
__global__ __launch_bounds__(256) void k_loss_rows(
    const float* __restrict__ ep, const float* __restrict__ ei,
    const float* __restrict__ ea, const float* __restrict__ tx,
    float* __restrict__ marg) {
  __shared__ float sp[256], si[256], sa[256];
  const int b = blockIdx.x, tid = threadIdx.x;
  const float* P = ep + (size_t)b * VIS2;
  const float* I = ei + (size_t)b * VIS2;
  const float* A = ea + (size_t)b * VIS2;
  const float* X = tx + (size_t)b * E_;
  float ap = 0.f, ai = 0.f, aa = 0.f;
  for (int k = tid; k < VIS2; k += 256) {
    float t = X[k];
    float d1 = P[k] - t; ap += d1 * d1;
    float d2 = I[k] - t; ai += d2 * d2;
    float d3 = A[k] - t; aa += d3 * d3;
  }
  sp[tid] = ap; si[tid] = ai; sa[tid] = aa;
  __syncthreads();
  for (int s = 128; s > 0; s >>= 1) {
    if (tid < s) { sp[tid] += sp[tid + s]; si[tid] += si[tid + s]; sa[tid] += sa[tid + s]; }
    __syncthreads();
  }
  if (tid == 0) {
    float m1 = sp[0] - si[0] + MARGIN;
    float m2 = sp[0] - sa[0] + MARGIN;
    marg[b]      = m1 > 0.f ? m1 : 0.f;
    marg[B_ + b] = m2 > 0.f ? m2 : 0.f;
  }
}

__global__ __launch_bounds__(256) void k_loss_final(const float* __restrict__ marg,
                                                    float* __restrict__ out) {
  __shared__ float s1[256], s2[256];
  const int tid = threadIdx.x;
  float a = 0.f, b = 0.f;
  for (int i = tid; i < B_; i += 256) { a += marg[i]; b += marg[B_ + i]; }
  s1[tid] = a; s2[tid] = b;
  __syncthreads();
  for (int s = 128; s > 0; s >>= 1) {
    if (tid < s) { s1[tid] += s1[tid + s]; s2[tid] += s2[tid + s]; }
    __syncthreads();
  }
  if (tid == 0) {
    float li = s1[0] / (float)B_ * LW_INTER;
    float la = s2[0] / (float)B_ * LW_INTRA;
    out[0] = li; out[1] = la; out[2] = li + la;
  }
}

// ---------------------------------------------------------------------------
// Host-side orchestration
// ---------------------------------------------------------------------------
extern "C" void kernel_launch(void* const* d_in, const int* in_sizes, int n_in,
                              void* d_out, int out_size, void* d_ws, size_t ws_size,
                              hipStream_t stream) {
  (void)in_sizes; (void)n_in; (void)out_size; (void)ws_size;

  const float* features_p = (const float*)d_in[0];
  const float* ts_p       = (const float*)d_in[1];
  const float* query      = (const float*)d_in[2];
  const float* cont       = (const float*)d_in[3];
  const float* feat_inter = (const float*)d_in[4];
  const float* feat_intra = (const float*)d_in[5];
  const float* ts_n       = (const float*)d_in[6];
  const float* W_v1 = (const float*)d_in[7];
  const float* b_v1 = (const float*)d_in[8];
  const float* W_v2 = (const float*)d_in[9];
  const float* b_v2 = (const float*)d_in[10];
  const float* W_ih = (const float*)d_in[11];
  const float* W_hh = (const float*)d_in[12];
  const float* b_ih = (const float*)d_in[13];
  const float* b_hh = (const float*)d_in[14];
  const float* W_le = (const float*)d_in[15];
  const float* b_le = (const float*)d_in[16];
  float* out = (float*)d_out;

  // ---- workspace carve (256B aligned) ----
  char* ws = (char*)d_ws;
  auto carve = [&](size_t bytes) -> void* {
    void* p = (void*)ws;
    ws += (bytes + 255) & ~(size_t)255;
    return p;
  };
  unsigned short* Wv1b = (unsigned short*)carve((size_t)VIS1 * KP1  * 2);
  unsigned short* Wv2b = (unsigned short*)carve((size_t)VIS2 * VIS1 * 2);
  unsigned short* Wcb  = (unsigned short*)carve((size_t)G4H  * KC   * 2);
  unsigned short* Wleb = (unsigned short*)carve((size_t)E_   * H_   * 2);
  unsigned short* Aimg = (unsigned short*)carve((size_t)B_   * KP1  * 2);
  unsigned short* hid1 = (unsigned short*)carve((size_t)B_   * VIS1 * 2);
  float* emb_p     = (float*)carve((size_t)B_ * VIS2 * 4);
  float* emb_inter = (float*)carve((size_t)B_ * VIS2 * 4);
  float* emb_intra = (float*)carve((size_t)B_ * VIS2 * 4);
  unsigned short* Al = (unsigned short*)carve((size_t)B_ * KC * 2);
  float* gates = (float*)carve((size_t)B_ * G4H * 4);
  float* cst   = (float*)carve((size_t)B_ * H_  * 4);
  float* hst   = (float*)carve((size_t)B_ * H_  * 4);
  float* text  = (float*)carve((size_t)B_ * E_  * 4);
  float* marg  = (float*)carve((size_t)2 * B_ * 4);

  const int TPB = 256;
  const int EW_BLOCKS = 4096;   // grid-stride elementwise kernels

  // ---- weight conversion (fp32 -> bf16, pad/fuse) ----
  k_convert_pad<<<EW_BLOCKS, TPB, 0, stream>>>(Wv1b, W_v1, VIS1, DV + 2, KP1);
  k_convert_pad<<<EW_BLOCKS, TPB, 0, stream>>>(Wv2b, W_v2, VIS2, VIS1, VIS1);
  k_convert_pad<<<EW_BLOCKS, TPB, 0, stream>>>(Wleb, W_le, E_, H_, H_);
  k_build_wc  <<<EW_BLOCKS, TPB, 0, stream>>>(Wcb, W_ih, W_hh);

  // ---- image model x3 ----
  const float* feats[3] = { features_p, feat_inter, feat_intra };
  const float* tss[3]   = { ts_p,       ts_p,       ts_n       };
  float* embs[3]        = { emb_p,      emb_inter,  emb_intra  };
  for (int i = 0; i < 3; ++i) {
    k_build_imgA<<<EW_BLOCKS, TPB, 0, stream>>>(Aimg, feats[i], tss[i]);
    // GEMM1: [B x KP1] * [VIS1 x KP1]^T + b_v1, ReLU -> bf16 hid1
    gemm_bf16_wmma<<<dim3(VIS1 / 64, B_ / 128), TPB, 0, stream>>>(
        Aimg, KP1, Wv1b, KP1, b_v1, nullptr, nullptr, hid1, VIS1, KP1, 1);
    // GEMM2: [B x VIS1] * [VIS2 x VIS1]^T + b_v2 -> fp32 emb
    gemm_bf16_wmma<<<dim3(VIS2 / 64, B_ / 128), TPB, 0, stream>>>(
        hid1, VIS1, Wv2b, VIS1, b_v2, nullptr, embs[i], nullptr, VIS2, VIS1, 0);
  }

  // ---- LSTM ----
  k_zero32<<<EW_BLOCKS, TPB, 0, stream>>>((unsigned int*)Al,  ((size_t)B_ * KC * 2) / 4);
  k_zero32<<<EW_BLOCKS, TPB, 0, stream>>>((unsigned int*)cst, ((size_t)B_ * H_ * 4) / 4);
  for (int t = 0; t < T_; ++t) {
    k_query_to_A<<<EW_BLOCKS, TPB, 0, stream>>>(Al, query + (size_t)t * B_ * DL);
    // gates = [x|h] @ [W_ih|W_hh]^T + b_ih + b_hh
    gemm_bf16_wmma<<<dim3(G4H / 64, B_ / 128), TPB, 0, stream>>>(
        Al, KC, Wcb, KC, b_ih, b_hh, gates, nullptr, G4H, KC, 0);
    const float* cont_next = (t + 1 < T_) ? (cont + (size_t)(t + 1) * B_) : nullptr;
    k_lstm_cell<<<EW_BLOCKS, TPB, 0, stream>>>(gates, cst, hst, Al, cont_next);
  }
  // text = h @ W_le^T + b_le  (h lives as bf16 in Al cols [DL, DL+H), lda=KC)
  gemm_bf16_wmma<<<dim3(E_ / 64, B_ / 128), TPB, 0, stream>>>(
      Al + DL, KC, Wleb, H_, b_le, nullptr, text, nullptr, E_, H_, 0);

  // ---- losses ----
  k_loss_rows<<<B_, TPB, 0, stream>>>(emb_p, emb_inter, emb_intra, text, marg);
  k_loss_final<<<1, TPB, 0, stream>>>(marg, out);
}
